// Seq2SeqRNN_1992864825890
// MI455X (gfx1250) — compile-verified, matrix operand order checked by reading, not run on hardware
//
#include <hip/hip_runtime.h>
#include <hip/hip_bf16.h>
#include <math.h>

typedef __attribute__((ext_vector_type(16))) _Float16 v16h;
typedef __attribute__((ext_vector_type(8)))  _Float16 v8h;
typedef __attribute__((ext_vector_type(8)))  float    v8f;

__device__ __forceinline__ float sigmoidf_(float x) { return 1.f / (1.f + __expf(-x)); }

__device__ __forceinline__ v16h mk16(v8h lo, v8h hi) {
    return __builtin_shufflevector(lo, hi, 0, 1, 2, 3, 4, 5, 6, 7,
                                   8, 9, 10, 11, 12, 13, 14, 15);
}

// ---------------------------------------------------------------------------
// WMMA GEMM, f16 operands, f32 accumulate:
//   C[M,N] (=|+=) A[M,K] @ W[K,N] (+bias0 +bias1)
// A: f16 row-major [M][lda]            (rows 16B-aligned, K%32==0, M%64==0)
// Wt: f16 TRANSPOSED row-major [Npad][ldwt] (Npad multiple of 64, zero padded)
// Per the ISA 16-bit A/B VGPR layouts, each lane's fragment is two contiguous
// 8-halfword runs in K -> two global_load_b128 per fragment, no LDS staging.
// Block: 128 threads = 4 waves; tile 64x64; wave = 16 rows x 64 cols
// (1 A frag reused against 4 B frags -> 4 v_wmma per 32-K chunk, no guards).
// ---------------------------------------------------------------------------
__global__ __launch_bounds__(128)
void gemm_f16_wmma(const _Float16* __restrict__ A, long lda,
                   const _Float16* __restrict__ Wt, long ldwt,
                   float* __restrict__ C, long ldc,
                   int M, int N, int K,
                   const float* __restrict__ bias0,
                   const float* __restrict__ bias1,
                   int accum)
{
    const int tid  = threadIdx.x;
    const int wave = tid >> 5;
    const int lane = tid & 31;
    const int half = lane >> 4;      // wave32: lanes 0-15 / 16-31
    const int mr   = lane & 15;
    const int bm = blockIdx.y * 64;
    const int bn = blockIdx.x * 64;

    const _Float16* __restrict__ arow = A  + (long)(bm + wave * 16 + mr) * lda;
    const _Float16* __restrict__ b0p  = Wt + (long)(bn +  0 + mr) * ldwt;
    const _Float16* __restrict__ b1p  = Wt + (long)(bn + 16 + mr) * ldwt;
    const _Float16* __restrict__ b2p  = Wt + (long)(bn + 32 + mr) * ldwt;
    const _Float16* __restrict__ b3p  = Wt + (long)(bn + 48 + mr) * ldwt;

    v8f vzero = {};
    v8f acc0 = vzero, acc1 = vzero, acc2 = vzero, acc3 = vzero;

    const int koff_lo = 8 * half;        // first 8-halfword run
    const int koff_hi = 16 + 8 * half;   // second 8-halfword run

    for (int k0 = 0; k0 < K; k0 += 32) {
        __builtin_prefetch(arow + k0 + koff_lo + 32, 0, 0);
        v16h af = mk16(*(const v8h*)(arow + k0 + koff_lo),
                       *(const v8h*)(arow + k0 + koff_hi));
        v16h bf0 = mk16(*(const v8h*)(b0p + k0 + koff_lo),
                        *(const v8h*)(b0p + k0 + koff_hi));
        v16h bf1 = mk16(*(const v8h*)(b1p + k0 + koff_lo),
                        *(const v8h*)(b1p + k0 + koff_hi));
        v16h bf2 = mk16(*(const v8h*)(b2p + k0 + koff_lo),
                        *(const v8h*)(b2p + k0 + koff_hi));
        v16h bf3 = mk16(*(const v8h*)(b3p + k0 + koff_lo),
                        *(const v8h*)(b3p + k0 + koff_hi));
        acc0 = __builtin_amdgcn_wmma_f32_16x16x32_f16(false, af, false, bf0, (short)0, acc0, false, false);
        acc1 = __builtin_amdgcn_wmma_f32_16x16x32_f16(false, af, false, bf1, (short)0, acc1, false, false);
        acc2 = __builtin_amdgcn_wmma_f32_16x16x32_f16(false, af, false, bf2, (short)0, acc2, false, false);
        acc3 = __builtin_amdgcn_wmma_f32_16x16x32_f16(false, af, false, bf3, (short)0, acc3, false, false);
    }

    // epilogue: 32-bit C/D layout (VGPR v holds rows v / v+8; cols = lane%16)
    v8f accs[4] = {acc0, acc1, acc2, acc3};
#pragma unroll
    for (int nt = 0; nt < 4; ++nt) {
        int col = bn + nt * 16 + mr;
        if (col >= N) continue;
        float badd = 0.f;
        if (bias0) badd += bias0[col];
        if (bias1) badd += bias1[col];
#pragma unroll
        for (int v = 0; v < 8; ++v) {
            int row = bm + wave * 16 + v + 8 * half;
            long idx = (long)row * ldc + col;
            float val = accs[nt][v] + badd;
            if (accum) val += C[idx];
            C[idx] = val;
        }
    }
}

// ---------------------------------------------------------------------------
// Conversion kernels (run once per launch; weights tiny vs. L2)
// ---------------------------------------------------------------------------
__global__ void fill_kernel(float* p, long n) {
    long i = (long)blockIdx.x * blockDim.x + threadIdx.x;
    if (i < n) p[i] = 0.f;
}

__global__ void copy_kernel(float* dst, const float* src, long n) {
    long i = (long)blockIdx.x * blockDim.x + threadIdx.x;
    if (i < n) dst[i] = src[i];
}

__global__ void convert_kernel(const float* __restrict__ src, _Float16* __restrict__ dst, long n) {
    long i = (long)blockIdx.x * blockDim.x + threadIdx.x;
    if (i < n) dst[i] = (_Float16)src[i];
}

// wt[n][k] = w[k][n], n < Npad (zero pad rows n >= N)
__global__ void convert_wT_kernel(const float* __restrict__ w, _Float16* __restrict__ wt,
                                  int K, int N, int Npad) {
    long i = (long)blockIdx.x * blockDim.x + threadIdx.x;
    if (i >= (long)Npad * K) return;
    int n = (int)(i / K), k = (int)(i % K);
    wt[i] = (n < N) ? (_Float16)w[(long)k * N + n] : (_Float16)0.f;
}

// fused decoder weight: wt[n][k], k<K1 from wih[k][N], else whh[k-K1][N]
__global__ void convert_wT2_kernel(const float* __restrict__ wih, const float* __restrict__ whh,
                                   _Float16* __restrict__ wt, int K1, int K2, int N) {
    int Kc = K1 + K2;
    long i = (long)blockIdx.x * blockDim.x + threadIdx.x;
    if (i >= (long)N * Kc) return;
    int n = (int)(i / Kc), k = (int)(i % Kc);
    float v = (k < K1) ? wih[(long)k * N + n] : whh[(long)(k - K1) * N + n];
    wt[i] = (_Float16)v;
}

// ---------------------------------------------------------------------------
// Elementwise kernels
// ---------------------------------------------------------------------------
// x[b,l,j] = in[b,l,0]*W[0,j] + in[b,l,1]*W[1,j] + bias[j]
__global__ void fc1_kernel(const float* __restrict__ in, const float* __restrict__ w,
                           const float* __restrict__ bias, float* __restrict__ x,
                           int HL, long total) {
    long idx = (long)blockIdx.x * blockDim.x + threadIdx.x;
    if (idx >= total) return;
    long bl = idx / HL;
    int  j  = (int)(idx % HL);
    x[idx] = in[bl * 2 + 0] * w[j] + in[bl * 2 + 1] * w[HL + j] + bias[j];
}

// BatchNorm over (B, HL) per sequence position l (block per l)
__global__ __launch_bounds__(256)
void bn_kernel(float* __restrict__ x, const float* __restrict__ gamma,
               const float* __restrict__ beta, int B, int L, int HL) {
    __shared__ float s1[256], s2[256];
    int l = blockIdx.x;
    int n = B * HL;
    float sum = 0.f, sq = 0.f;
    for (int i = threadIdx.x; i < n; i += 256) {
        int b = i / HL, j = i % HL;
        float v = x[(long)b * L * HL + (long)l * HL + j];
        sum += v; sq += v * v;
    }
    s1[threadIdx.x] = sum; s2[threadIdx.x] = sq;
    __syncthreads();
    for (int st = 128; st > 0; st >>= 1) {
        if (threadIdx.x < st) { s1[threadIdx.x] += s1[threadIdx.x + st]; s2[threadIdx.x] += s2[threadIdx.x + st]; }
        __syncthreads();
    }
    float mu  = s1[0] / n;
    float var = s2[0] / n - mu * mu;
    float inv = rsqrtf(var + 1e-5f);
    float g = gamma[l], bb = beta[l];
    for (int i = threadIdx.x; i < n; i += 256) {
        int b = i / HL, j = i % HL;
        long idx = (long)b * L * HL + (long)l * HL + j;
        x[idx] = g * (x[idx] - mu) * inv + bb;
    }
}

// gates [B,4H] (i,f,g,o); c (f32) updated in place; h written as f16 to one
// or two destinations (encoder: encOut16 slice; decoder: hcat16 half + xh tail)
__global__ void lstm_cell_kernel(const float* __restrict__ gates, float* __restrict__ c,
                                 _Float16* __restrict__ h0, long s0,
                                 _Float16* __restrict__ h1, long s1, int H) {
    int idx = blockIdx.x * blockDim.x + threadIdx.x;  // B*H
    int b = idx / H, j = idx % H;
    long gb = (long)b * 4 * H;
    float gi = gates[gb + j];
    float gf = gates[gb + H + j];
    float gg = gates[gb + 2 * H + j];
    float go = gates[gb + 3 * H + j];
    float cn = sigmoidf_(gf) * c[idx] + sigmoidf_(gi) * tanhf(gg);
    c[idx] = cn;
    _Float16 h = (_Float16)(sigmoidf_(go) * tanhf(cn));
    h0[(long)b * s0 + j] = h;
    if (h1) h1[(long)b * s1 + j] = h;
}

// hcat16 = [hf_final | hb_final]; also seed decoder xh tails with h0
__global__ void hcat_init_kernel(const _Float16* __restrict__ encOut16,
                                 _Float16* __restrict__ hcat,
                                 _Float16* __restrict__ xhf, _Float16* __restrict__ xhb,
                                 int L, int H, int XW) {
    int idx = blockIdx.x * blockDim.x + threadIdx.x;  // B*2H
    int b = idx / (2 * H), j = idx % (2 * H);
    long base = (long)b * L * 2 * H;
    _Float16 v = (j < H) ? encOut16[base + (long)(L - 1) * 2 * H + j]
                         : encOut16[base + j];
    hcat[(long)b * 2 * H + j] = v;
    if (j < H) xhf[(long)b * XW + 1024 + j] = v;
    else       xhb[(long)b * XW + 1024 + (j - H)] = v;
}

// u[b,l] = sum_q tanh(Wenc[b,l,q] + Vh[b,q]) * vw[q] + vb   (one wave per (b,l))
__global__ __launch_bounds__(256)
void attn_u_kernel(const float* __restrict__ Wenc, const float* __restrict__ Vh,
                   const float* __restrict__ vw, const float* __restrict__ vb,
                   float* __restrict__ u, int BL, int L, int SQ) {
    int wid  = blockIdx.x * 8 + (threadIdx.x >> 5);
    int lane = threadIdx.x & 31;
    if (wid >= BL) return;
    int b = wid / L;
    float s = 0.f;
    for (int q = lane; q < SQ; q += 32)
        s += tanhf(Wenc[(long)wid * SQ + q] + Vh[(long)b * SQ + q]) * vw[q];
    for (int off = 16; off > 0; off >>= 1) s += __shfl_xor(s, off, 32);
    if (lane == 0) u[wid] = s + vb[0];
}

__global__ __launch_bounds__(128)
void softmax_kernel(const float* __restrict__ u, float* __restrict__ attn, int L) {
    __shared__ float sh[128];
    int b = blockIdx.x, t = threadIdx.x;
    float v = (t < L) ? u[(long)b * L + t] : -1e30f;
    sh[t] = v; __syncthreads();
    for (int st = 64; st > 0; st >>= 1) { if (t < st) sh[t] = fmaxf(sh[t], sh[t + st]); __syncthreads(); }
    float mx = sh[0]; __syncthreads();
    float e = (t < L) ? __expf(v - mx) : 0.f;
    sh[t] = e; __syncthreads();
    for (int st = 64; st > 0; st >>= 1) { if (t < st) sh[t] += sh[t + st]; __syncthreads(); }
    if (t < L) attn[(long)b * L + t] = e / sh[0];
}

// xh[b, 0:2H] = din[b]*fc1w + fc1b ; xh[b, 2H:4H] = sum_l encOut16[b,l,:]*attn[b,l]
// written identically into both direction buffers (h tails untouched)
__global__ __launch_bounds__(256)
void ctx_xx_kernel(const _Float16* __restrict__ encOut16, const float* __restrict__ attn,
                   const float* __restrict__ din, const float* __restrict__ w,
                   const float* __restrict__ bias,
                   _Float16* __restrict__ xhf, _Float16* __restrict__ xhb,
                   int L, int H2, int XW) {
    int b = blockIdx.x, t = threadIdx.x;
    float d = din[b];
    for (int j = t; j < H2; j += 256) {
        _Float16 v = (_Float16)(d * w[j] + bias[j]);
        xhf[(long)b * XW + j] = v;
        xhb[(long)b * XW + j] = v;
    }
    for (int j = t; j < H2; j += 256) {
        const _Float16* eb = encOut16 + (long)b * L * H2 + j;
        const float* ab = attn + (long)b * L;
        float s = 0.f;
        for (int l = 0; l < L; ++l) s += (float)eb[(long)l * H2] * ab[l];
        _Float16 v = (_Float16)s;
        xhf[(long)b * XW + H2 + j] = v;
        xhb[(long)b * XW + H2 + j] = v;
    }
}

// dout[b] = softplus(out2[b,:].fc3w + fc3b); also dec_outs[b,step]
__global__ __launch_bounds__(128)
void fc3_kernel(const float* __restrict__ out2, const float* __restrict__ w,
                const float* __restrict__ bias, float* __restrict__ dout,
                float* __restrict__ dec_outs, int step, int H, int L) {
    __shared__ float sh[128];
    int b = blockIdx.x, t = threadIdx.x;
    float s = 0.f;
    for (int j = t; j < H; j += 128) s += out2[(long)b * H + j] * w[j];
    sh[t] = s; __syncthreads();
    for (int st = 64; st > 0; st >>= 1) { if (t < st) sh[t] += sh[t + st]; __syncthreads(); }
    if (t == 0) {
        float v = sh[0] + bias[0];
        float sp = (v > 20.f) ? v : log1pf(__expf(v));
        dout[b] = sp;
        dec_outs[(long)b * L + step] = sp;
    }
}

// t = cumsum/max; Bernstein degree-3 basis
__global__ void t_bernstein_kernel(const float* __restrict__ dec_outs, float* __restrict__ t_out,
                                   float* __restrict__ A, int B, int L) {
    int b = blockIdx.x * blockDim.x + threadIdx.x;
    if (b >= B) return;
    float tot = 0.f;
    for (int l = 0; l < L; ++l) tot += dec_outs[(long)b * L + l];
    float inv = 1.f / tot;   // softplus > 0 -> tot > 0
    float acc = 0.f;
    for (int l = 0; l < L; ++l) {
        acc += dec_outs[(long)b * L + l];
        float t = acc * inv, uo = 1.f - t;
        t_out[(long)b * L + l] = t;
        long base = ((long)b * L + l) * 4;
        A[base + 0] = uo * uo * uo;
        A[base + 1] = 3.f * t * uo * uo;
        A[base + 2] = 3.f * t * t * uo;
        A[base + 3] = t * t * t;
    }
}

// Per-batch least squares on rows 1..L-2 (normal equations, 4x4 solve) + eval
__global__ __launch_bounds__(128)
void spline_fit_kernel(const float* __restrict__ A, const float* __restrict__ curves,
                       float* __restrict__ splines, int L) {
    __shared__ float red[18][128];
    __shared__ float cc[8];
    int b = blockIdx.x, t = threadIdx.x;
    float a[4] = {0, 0, 0, 0}, y[2] = {0, 0};
    int l = 1 + t;
    if (t < L - 2) {
        for (int k = 0; k < 4; ++k) a[k] = A[((long)b * L + l) * 4 + k];
        y[0] = curves[((long)b * L + l) * 2 + 0];
        y[1] = curves[((long)b * L + l) * 2 + 1];
    }
    int idx = 0;
    for (int i = 0; i < 4; ++i)
        for (int j = i; j < 4; ++j) red[idx++][t] = a[i] * a[j];   // 10 of G
    for (int i = 0; i < 4; ++i)
        for (int p = 0; p < 2; ++p) red[idx++][t] = a[i] * y[p];   // 8 of R
    __syncthreads();
    for (int st = 64; st > 0; st >>= 1) {
        if (t < st)
            for (int r = 0; r < 18; ++r) red[r][t] += red[r][t + st];
        __syncthreads();
    }
    if (t == 0) {
        float M_[4][6];
        int q = 0;
        for (int i = 0; i < 4; ++i)
            for (int j = i; j < 4; ++j) { M_[i][j] = red[q][0]; M_[j][i] = red[q][0]; ++q; }
        for (int i = 0; i < 4; ++i) { M_[i][4] = red[10 + i * 2][0]; M_[i][5] = red[10 + i * 2 + 1][0]; }
        for (int col = 0; col < 4; ++col) {
            int piv = col; float best = fabsf(M_[col][col]);
            for (int r = col + 1; r < 4; ++r) {
                float v = fabsf(M_[r][col]);
                if (v > best) { best = v; piv = r; }
            }
            if (piv != col)
                for (int c2 = 0; c2 < 6; ++c2) { float tmp = M_[col][c2]; M_[col][c2] = M_[piv][c2]; M_[piv][c2] = tmp; }
            float d = M_[col][col];
            if (fabsf(d) < 1e-20f) d = 1e-20f;
            float invd = 1.f / d;
            for (int c2 = col; c2 < 6; ++c2) M_[col][c2] *= invd;
            for (int r = 0; r < 4; ++r)
                if (r != col) {
                    float f = M_[r][col];
                    for (int c2 = col; c2 < 6; ++c2) M_[r][c2] -= f * M_[col][c2];
                }
        }
        for (int k = 0; k < 4; ++k) { cc[k * 2] = M_[k][4]; cc[k * 2 + 1] = M_[k][5]; }
    }
    __syncthreads();
    for (int l2 = t; l2 < L; l2 += 128) {
        float s0 = 0.f, s1 = 0.f;
        for (int k = 0; k < 4; ++k) {
            float av = A[((long)b * L + l2) * 4 + k];
            s0 += av * cc[k * 2];
            s1 += av * cc[k * 2 + 1];
        }
        splines[((long)b * L + l2) * 2 + 0] = s0;
        splines[((long)b * L + l2) * 2 + 1] = s1;
    }
}

// ---------------------------------------------------------------------------
// Host side
// ---------------------------------------------------------------------------
static inline void gemm(hipStream_t s, float* C, long ldc,
                        const _Float16* A, long lda,
                        const _Float16* Wt, long ldwt,
                        int M, int N, int K,
                        const float* b0, const float* b1, int accum) {
    dim3 grid((N + 63) / 64, (M + 63) / 64);
    gemm_f16_wmma<<<grid, 128, 0, s>>>(A, lda, Wt, ldwt, C, ldc, M, N, K, b0, b1, accum);
}

extern "C" void kernel_launch(void* const* d_in, const int* in_sizes, int n_in,
                              void* d_out, int out_size, void* d_ws, size_t ws_size,
                              hipStream_t stream) {
    (void)in_sizes; (void)n_in; (void)out_size; (void)ws_size;
    const int B = 512, L = 100, HL = 256, H = 256, SQ = 100;
    const int H2 = 2 * H, H4 = 4 * H;
    const int XW = H4 + H;            // 1280 : [inp(512) | ctx(512) | h(256)]

    const float* input     = (const float*)d_in[0];
    const float* enc_fc1_w = (const float*)d_in[1];
    const float* enc_fc1_b = (const float*)d_in[2];
    const float* bn_gamma  = (const float*)d_in[3];
    const float* bn_beta   = (const float*)d_in[4];
    const float* enc_wih_f = (const float*)d_in[5];
    const float* enc_whh_f = (const float*)d_in[6];
    const float* enc_bih_f = (const float*)d_in[7];
    const float* enc_bhh_f = (const float*)d_in[8];
    const float* enc_wih_b = (const float*)d_in[9];
    const float* enc_whh_b = (const float*)d_in[10];
    const float* enc_bih_b = (const float*)d_in[11];
    const float* enc_bhh_b = (const float*)d_in[12];
    const float* dec_W_w   = (const float*)d_in[13];
    const float* dec_W_b   = (const float*)d_in[14];
    const float* dec_V_w   = (const float*)d_in[15];
    const float* dec_V_b   = (const float*)d_in[16];
    const float* dec_v_w   = (const float*)d_in[17];
    const float* dec_v_b   = (const float*)d_in[18];
    const float* dec_fc1_w = (const float*)d_in[19];
    const float* dec_fc1_b = (const float*)d_in[20];
    const float* dec_wih_f = (const float*)d_in[21];
    const float* dec_whh_f = (const float*)d_in[22];
    const float* dec_bih_f = (const float*)d_in[23];
    const float* dec_bhh_f = (const float*)d_in[24];
    const float* dec_wih_b = (const float*)d_in[25];
    const float* dec_whh_b = (const float*)d_in[26];
    const float* dec_bih_b = (const float*)d_in[27];
    const float* dec_bhh_b = (const float*)d_in[28];
    const float* fc2_w     = (const float*)d_in[29];
    const float* fc2_b     = (const float*)d_in[30];
    const float* fc3_w     = (const float*)d_in[31];
    const float* fc3_b     = (const float*)d_in[32];

    // ---------------- f32 workspace ----------------
    float* ws = (float*)d_ws;
    float* x        = ws;                               // B*L*HL  = 13,107,200
    float* Wenc     = x + (long)B * L * HL;             // B*L*SQ  =  5,120,000
    float* gates_f  = Wenc + (long)B * L * SQ;          // B*4H
    float* gates_b  = gates_f + (long)B * H4;
    float* cf       = gates_b + (long)B * H4;           // B*H
    float* cb       = cf + (long)B * H;
    float* Vh       = cb + (long)B * H;                 // B*SQ
    float* uu       = Vh + (long)B * SQ;                // B*L
    float* attn     = uu + (long)B * L;                 // B*L
    float* out2     = attn + (long)B * L;               // B*H
    float* doutb    = out2 + (long)B * H;               // B
    float* dec_outs = doutb + B;                        // B*L
    float* Acoef    = dec_outs + (long)B * L;           // B*L*4
    float* f32_end  = Acoef + (long)B * L * 4;

    // ---------------- f16 workspace (16B aligned sub-buffers) ----------------
    long f32_total = (long)(f32_end - ws);
    f32_total = (f32_total + 7) & ~7L;
    _Float16* h16 = (_Float16*)(ws + f32_total);
    _Float16* x16       = h16;                          // B*L*HL
    _Float16* encOut16  = x16 + (long)B * L * HL;       // B*L*2H
    _Float16* hcat16    = encOut16 + (long)B * L * H2;  // B*2H
    _Float16* xhf16     = hcat16 + (long)B * H2;        // B*XW
    _Float16* xhb16     = xhf16 + (long)B * XW;         // B*XW
    _Float16* wT_ewih_f = xhb16 + (long)B * XW;         // [1024][256]
    _Float16* wT_ewhh_f = wT_ewih_f + (long)H4 * HL;    // [1024][256]
    _Float16* wT_ewih_b = wT_ewhh_f + (long)H4 * H;
    _Float16* wT_ewhh_b = wT_ewih_b + (long)H4 * HL;
    _Float16* wT_decW   = wT_ewhh_b + (long)H4 * H;     // [128][512]
    _Float16* wT_decV   = wT_decW + (long)128 * H2;     // [128][512]
    _Float16* wT_dcmb_f = wT_decV + (long)128 * H2;     // [1024][1280]
    _Float16* wT_dcmb_b = wT_dcmb_f + (long)H4 * XW;
    _Float16* wT_fc2    = wT_dcmb_b + (long)H4 * XW;    // [256][512]

    float* splines_out = (float*)d_out;
    float* t_out       = splines_out + (long)B * L * 2;
    float* curves_out  = t_out + (long)B * L;

    // ---- zero init: cf|cb contiguous, doutb|dec_outs contiguous ----
    {
        long n1 = (long)2 * B * H;
        fill_kernel<<<(int)((n1 + 255) / 256), 256, 0, stream>>>(cf, n1);
        long n2 = (long)B + (long)B * L;
        fill_kernel<<<(int)((n2 + 255) / 256), 256, 0, stream>>>(doutb, n2);
    }

    // ---- weight conversion (f32 -> f16, transposed, padded) ----
    {
        long n = (long)H4 * HL;   // 1024*256
        int g = (int)((n + 255) / 256);
        convert_wT_kernel<<<g, 256, 0, stream>>>(enc_wih_f, wT_ewih_f, HL, H4, H4);
        convert_wT_kernel<<<g, 256, 0, stream>>>(enc_whh_f, wT_ewhh_f, H,  H4, H4);
        convert_wT_kernel<<<g, 256, 0, stream>>>(enc_wih_b, wT_ewih_b, HL, H4, H4);
        convert_wT_kernel<<<g, 256, 0, stream>>>(enc_whh_b, wT_ewhh_b, H,  H4, H4);
        long nw = (long)128 * H2;
        convert_wT_kernel<<<(int)((nw + 255) / 256), 256, 0, stream>>>(dec_W_w, wT_decW, H2, SQ, 128);
        convert_wT_kernel<<<(int)((nw + 255) / 256), 256, 0, stream>>>(dec_V_w, wT_decV, H2, SQ, 128);
        long nc = (long)H4 * XW;  // 1024*1280
        convert_wT2_kernel<<<(int)((nc + 255) / 256), 256, 0, stream>>>(dec_wih_f, dec_whh_f, wT_dcmb_f, H4, H, H4);
        convert_wT2_kernel<<<(int)((nc + 255) / 256), 256, 0, stream>>>(dec_wih_b, dec_whh_b, wT_dcmb_b, H4, H, H4);
        long nf = (long)H * H2;   // fc2: [512][256] -> [256][512]
        convert_wT_kernel<<<(int)((nf + 255) / 256), 256, 0, stream>>>(fc2_w, wT_fc2, H2, H, H);
    }

    // ---- encoder fc1 + BN + convert activations ----
    {
        long total = (long)B * L * HL;
        fc1_kernel<<<(int)((total + 255) / 256), 256, 0, stream>>>(input, enc_fc1_w, enc_fc1_b, x, HL, total);
        bn_kernel<<<L, 256, 0, stream>>>(x, bn_gamma, bn_beta, B, L, HL);
        convert_kernel<<<(int)((total + 255) / 256), 256, 0, stream>>>(x, x16, total);
    }

    // ---- encoder biLSTM (h_{t-1} read from encOut16 via row strides) ----
    const long xStride  = (long)L * HL;   // per-b stride in x16
    const long eoStride = (long)L * H2;   // per-b stride in encOut16
    for (int s = 0; s < L; ++s) {
        // forward
        gemm(stream, gates_f, H4, x16 + (long)s * HL, xStride, wT_ewih_f, HL,
             B, H4, HL, enc_bih_f, enc_bhh_f, 0);
        if (s > 0)
            gemm(stream, gates_f, H4, encOut16 + (long)(s - 1) * H2, eoStride, wT_ewhh_f, H,
                 B, H4, H, nullptr, nullptr, 1);
        lstm_cell_kernel<<<(B * H) / 256, 256, 0, stream>>>(
            gates_f, cf, encOut16 + (long)s * H2, eoStride, (_Float16*)nullptr, 0, H);
        // backward
        int sb = L - 1 - s;
        gemm(stream, gates_b, H4, x16 + (long)sb * HL, xStride, wT_ewih_b, HL,
             B, H4, HL, enc_bih_b, enc_bhh_b, 0);
        if (s > 0)
            gemm(stream, gates_b, H4, encOut16 + (long)(sb + 1) * H2 + H, eoStride, wT_ewhh_b, H,
                 B, H4, H, nullptr, nullptr, 1);
        lstm_cell_kernel<<<(B * H) / 256, 256, 0, stream>>>(
            gates_b, cb, encOut16 + (long)sb * H2 + H, eoStride, (_Float16*)nullptr, 0, H);
    }

    // ---- Wenc = encOut @ dec_W_w + dec_W_b  (one big GEMM, M = B*L) ----
    gemm(stream, Wenc, SQ, encOut16, H2, wT_decW, H2, B * L, SQ, H2, dec_W_b, nullptr, 0);

    // ---- decoder init: hcat16 + xh tails; cf/cb already hold final states ----
    hcat_init_kernel<<<(B * H2) / 256, 256, 0, stream>>>(encOut16, hcat16, xhf16, xhb16, L, H, XW);

    // ---- decoder loop (99 sequential steps) ----
    for (int s = 1; s < L; ++s) {
        gemm(stream, Vh, SQ, hcat16, H2, wT_decV, H2, B, SQ, H2, dec_V_b, nullptr, 0);
        attn_u_kernel<<<(B * L) / 8, 256, 0, stream>>>(Wenc, Vh, dec_v_w, dec_v_b, uu, B * L, L, SQ);
        softmax_kernel<<<B, 128, 0, stream>>>(uu, attn, L);
        ctx_xx_kernel<<<B, 256, 0, stream>>>(encOut16, attn, doutb, dec_fc1_w, dec_fc1_b,
                                             xhf16, xhb16, L, H2, XW);
        // fused gates = [inp|ctx|h] @ [wih;whh]  (reads old h in xh tails)
        gemm(stream, gates_f, H4, xhf16, XW, wT_dcmb_f, XW, B, H4, XW, dec_bih_f, dec_bhh_f, 0);
        gemm(stream, gates_b, H4, xhb16, XW, wT_dcmb_b, XW, B, H4, XW, dec_bih_b, dec_bhh_b, 0);
        lstm_cell_kernel<<<(B * H) / 256, 256, 0, stream>>>(
            gates_f, cf, hcat16, (long)H2, xhf16 + 1024, (long)XW, H);
        lstm_cell_kernel<<<(B * H) / 256, 256, 0, stream>>>(
            gates_b, cb, hcat16 + H, (long)H2, xhb16 + 1024, (long)XW, H);
        // output head
        gemm(stream, out2, H, hcat16, H2, wT_fc2, H2, B, H, H2, fc2_b, nullptr, 0);
        fc3_kernel<<<B, 128, 0, stream>>>(out2, fc3_w, fc3_b, doutb, dec_outs, s, H, L);
    }

    // ---- parameterization, Bernstein basis, spline fit, outputs ----
    t_bernstein_kernel<<<(B + 127) / 128, 128, 0, stream>>>(dec_outs, t_out, Acoef, B, L);
    spline_fit_kernel<<<B, 128, 0, stream>>>(Acoef, input, splines_out, L);
    copy_kernel<<<(int)(((long)B * L * 2 + 255) / 256), 256, 0, stream>>>(curves_out, input, (long)B * L * 2);
}
